// MaskedMultiSelfAttention_46763603918977
// MI455X (gfx1250) — matrix operand order repaired and checked by reading, not compiled, not run on hardware
//
#include <hip/hip_runtime.h>
#include <hip/hip_bf16.h>

// ---------------------------------------------------------------------------
// Masked multi-head self-attention for MI455X (gfx1250, wave32, WMMA).
// f16 operands + f32 accumulation via v_wmma_f32_16x16x32_f16.
// Weights/activations pre-converted to f16 (weights pre-transposed) so GEMM
// inner loops are pure {global_load_b128 (A), async-to-LDS (B), wmma}.
// ---------------------------------------------------------------------------

typedef __attribute__((ext_vector_type(16))) _Float16 v16h;
typedef __attribute__((ext_vector_type(8)))  _Float16 v8h;
typedef __attribute__((ext_vector_type(4)))  _Float16 v4h;
typedef __attribute__((ext_vector_type(8)))  float    v8f;
typedef __attribute__((ext_vector_type(4)))  float    vf4;

#define C_DIM   1024
#define T_DIM   2048
#define N_HEADS 16
#define D_HEAD  64
#define B_DIM   4
#define THREECC 3072

static __device__ __forceinline__ v16h cat16(v8h lo, v8h hi) {
    return __builtin_shufflevector(lo, hi, 0,1,2,3,4,5,6,7,8,9,10,11,12,13,14,15);
}
static __device__ __forceinline__ v8f v8f_zero() {
    v8f z;
#pragma unroll
    for (int i = 0; i < 8; ++i) z[i] = 0.0f;
    return z;
}
static __device__ __forceinline__ v8f wmma_f16(v16h a, v16h b, v8f c) {
    return __builtin_amdgcn_wmma_f32_16x16x32_f16(false, a, false, b, (short)0, c, false, false);
}
static __device__ __forceinline__ void async_copy16(unsigned int lds_off,
                                                    const _Float16* gptr) {
    // GLOBAL_LOAD_ASYNC_TO_LDS_B128: per-lane 16B global -> LDS, ASYNCcnt-tracked
    asm volatile("global_load_async_to_lds_b128 %0, %1, off"
                 :: "v"(lds_off), "v"(gptr) : "memory");
}
// s_wait_asynccnt needs an immediate operand -> constant wrappers.
static __device__ __forceinline__ void wait_async_4() {
#if __has_builtin(__builtin_amdgcn_s_wait_asynccnt)
    __builtin_amdgcn_s_wait_asynccnt(4);
#else
    asm volatile("s_wait_asynccnt 0x4" ::: "memory");
#endif
}
static __device__ __forceinline__ void wait_async_0() {
#if __has_builtin(__builtin_amdgcn_s_wait_asynccnt)
    __builtin_amdgcn_s_wait_asynccnt(0);
#else
    asm volatile("s_wait_asynccnt 0x0" ::: "memory");
#endif
}

// ---------------------------------------------------------------------------
// Pre-pass 0a: x (f32) -> xh (f16), elementwise. 4 elems/thread.
// ---------------------------------------------------------------------------
__global__ void cvt_x_kernel(const float* __restrict__ x, _Float16* __restrict__ xh) {
    const size_t i = ((size_t)blockIdx.x * 256 + threadIdx.x) * 4;
    vf4 v = *(const vf4*)(x + i);
    v4h h;
    h[0] = (_Float16)v[0]; h[1] = (_Float16)v[1];
    h[2] = (_Float16)v[2]; h[3] = (_Float16)v[3];
    *(v4h*)(xh + i) = h;
}

// ---------------------------------------------------------------------------
// Pre-pass 0b: W [1024][ncols] f32 -> Wt [ncols][1024] f16 (LDS tile transpose,
// both global accesses coalesced).
// ---------------------------------------------------------------------------
__global__ void cvt_transpose_kernel(const float* __restrict__ W,
                                     _Float16* __restrict__ Wt, int ncols) {
    __shared__ _Float16 sT[32 * 33];
    const int n0 = blockIdx.x * 32, k0 = blockIdx.y * 32;
    const int tx = threadIdx.x & 31, ty = threadIdx.x >> 5;   // 32 x 8
#pragma unroll
    for (int p = 0; p < 4; ++p) {
        const int k = p * 8 + ty;
        sT[k * 33 + tx] = (_Float16)W[(size_t)(k0 + k) * ncols + n0 + tx];
    }
    __syncthreads();
#pragma unroll
    for (int p = 0; p < 4; ++p) {
        const int n = p * 8 + ty;
        Wt[(size_t)(n0 + n) * C_DIM + k0 + tx] = sT[tx * 33 + n];
    }
}

// ---------------------------------------------------------------------------
// Kernel 1: qkv = xh @ Wat^T + b, scattered to per-head f16 buffers.
//   Q -> [b,h,t,d] (scaled 1/sqrt(D)),  K -> [b,h,t,d],  V -> [b,h,d,t]
// WG = 128 (4 waves, 2x2), WG tile 64M x 128N, K step 32.
// A: direct global (WMMA A-layout). B: async-to-LDS, double-buffered.
// ---------------------------------------------------------------------------
__global__ void qkv_gemm_kernel(const _Float16* __restrict__ xh,
                                const _Float16* __restrict__ Wt,   // [3072][1024]
                                const float* __restrict__ bias,
                                _Float16* __restrict__ Qh,
                                _Float16* __restrict__ Kh,
                                _Float16* __restrict__ Vh) {
    const int tid  = threadIdx.x;
    const int lane = tid & 31;
    const int wave = tid >> 5;
    const int wm   = wave >> 1;
    const int wn   = wave & 1;
    const int lrow = lane & 15;
    const int hi   = (lane >> 4) & 1;

    const int mbase = blockIdx.y * 64;
    const int nbase = blockIdx.x * 128;

    __shared__ __align__(16) _Float16 sB[2][128 * 32];   // [n][k] f16, 8KB/buf
    const unsigned int sb_base = (unsigned int)(uintptr_t)&sB[0][0];

    v8f c[2][4];
#pragma unroll
    for (int mi = 0; mi < 2; ++mi)
#pragma unroll
        for (int ni = 0; ni < 4; ++ni) c[mi][ni] = v8f_zero();

    // issue 4 async 16B copies per thread = one 128n x 32k f16 tile per WG
    auto issue_tile = [&](int kt, int buf) {
#pragma unroll
        for (int j = 0; j < 4; ++j) {
            const int cidx = j * 128 + tid;        // 0..511 chunks of 16B
            const int n    = cidx >> 2;
            const int kh   = cidx & 3;
            const unsigned int lo = sb_base +
                (unsigned int)(buf * (128 * 32 * 2) + n * 64 + kh * 16);
            const _Float16* gp = Wt + ((size_t)(nbase + n) << 10) + kt * 32 + kh * 8;
            async_copy16(lo, gp);
        }
    };

    issue_tile(0, 0);

    for (int kt = 0; kt < C_DIM / 32; ++kt) {
        const int cur = kt & 1;
        __syncthreads();                       // everyone done reading buf[1-cur]
        const bool more = (kt + 1) < (C_DIM / 32);
        if (more) {
            issue_tile(kt + 1, 1 - cur);
            wait_async_4();                    // tile kt landed (4 newer pending)
        } else {
            wait_async_0();
        }
        __syncthreads();                       // whole tile kt visible WG-wide

        const int k0 = kt * 32;
        // A operands straight from global in WMMA A-layout
        v16h a[2];
#pragma unroll
        for (int mi = 0; mi < 2; ++mi) {
            const int row = mbase + wm * 32 + mi * 16 + lrow;
            const int ko  = hi * 16;
            v8h lo = *(const v8h*)(xh + (size_t)row * C_DIM + k0 + ko);
            v8h hh = *(const v8h*)(xh + (size_t)row * C_DIM + k0 + ko + 8);
            a[mi] = cat16(lo, hh);
        }
#pragma unroll
        for (int ni = 0; ni < 4; ++ni) {
            const int nn = wn * 64 + ni * 16 + lrow;
            const int ko = hi * 16;
            v8h lo = *(const v8h*)(&sB[cur][nn * 32 + ko]);
            v8h hh = *(const v8h*)(&sB[cur][nn * 32 + ko + 8]);
            v16h bb = cat16(lo, hh);
#pragma unroll
            for (int mi = 0; mi < 2; ++mi)
                c[mi][ni] = wmma_f16(a[mi], bb, c[mi][ni]);
        }
    }

    // epilogue: bias + scatter into per-head f16 layouts
#pragma unroll
    for (int ni = 0; ni < 4; ++ni) {
        const int gn = nbase + wn * 64 + ni * 16 + lrow;
        const float bv = bias[gn];
        const int sec = gn >> 10;
        const int cc  = gn & 1023;
        const int h   = cc >> 6;
        const int d   = cc & 63;
#pragma unroll
        for (int mi = 0; mi < 2; ++mi)
#pragma unroll
            for (int r = 0; r < 8; ++r) {
                const int gm = mbase + wm * 32 + mi * 16 + r + hi * 8;
                const int b  = gm >> 11;
                const int t  = gm & 2047;
                const float val = c[mi][ni][r] + bv;
                const size_t bh = (size_t)(b * N_HEADS + h);
                if (sec == 0)
                    Qh[(bh * T_DIM + t) * D_HEAD + d] = (_Float16)(val * 0.125f);
                else if (sec == 1)
                    Kh[(bh * T_DIM + t) * D_HEAD + d] = (_Float16)val;
                else
                    Vh[(bh * D_HEAD + d) * T_DIM + t] = (_Float16)val;
            }
    }
}

// ---------------------------------------------------------------------------
// Kernel 2: causal flash attention. WG = 128 (4 waves); each wave owns 16
// query rows of one (b,h); 32-key tiles; online softmax; f16 out [b,t,h*64+d].
// ---------------------------------------------------------------------------
__global__ void flash_attn_kernel(const _Float16* __restrict__ Qh,
                                  const _Float16* __restrict__ Kh,
                                  const _Float16* __restrict__ Vh,
                                  _Float16* __restrict__ AO) {
    const int tid  = threadIdx.x;
    const int lane = tid & 31;
    const int wave = tid >> 5;
    const int lrow = lane & 15;
    const int hi   = (lane >> 4) & 1;

    const int bh = blockIdx.y;
    const int b  = bh >> 4;
    const int h  = bh & 15;
    const int qbase = blockIdx.x * 64 + wave * 16;

    const _Float16* Qb = Qh + (size_t)bh * T_DIM * D_HEAD;
    const _Float16* Kb = Kh + (size_t)bh * T_DIM * D_HEAD;
    const _Float16* Vb = Vh + (size_t)bh * D_HEAD * T_DIM;

    __shared__ __align__(16) _Float16 sP[4][16 * 32];

    v16h aQ[2];
    {
        const int row = qbase + lrow;
        const int ko  = hi * 16;
#pragma unroll
        for (int dc = 0; dc < 2; ++dc) {
            v8h lo = *(const v8h*)(Qb + (size_t)row * D_HEAD + dc * 32 + ko);
            v8h hh = *(const v8h*)(Qb + (size_t)row * D_HEAD + dc * 32 + ko + 8);
            aQ[dc] = cat16(lo, hh);
        }
    }

    float m_i[8], l_i[8];
    v8f o[4];
#pragma unroll
    for (int r = 0; r < 8; ++r) { m_i[r] = -3.0e38f; l_i[r] = 0.0f; }
#pragma unroll
    for (int dn = 0; dn < 4; ++dn) o[dn] = v8f_zero();

    _Float16* sp = sP[wave];
    const int kend = qbase + 16;

    for (int k0 = 0; k0 < kend; k0 += 32) {
        v8f s[2];
#pragma unroll
        for (int kn = 0; kn < 2; ++kn) {
            s[kn] = v8f_zero();
            const int key = k0 + kn * 16 + lrow;
            const int ko  = hi * 16;
#pragma unroll
            for (int dc = 0; dc < 2; ++dc) {
                v8h lo = *(const v8h*)(Kb + (size_t)key * D_HEAD + dc * 32 + ko);
                v8h hh = *(const v8h*)(Kb + (size_t)key * D_HEAD + dc * 32 + ko + 8);
                s[kn] = wmma_f16(aQ[dc], cat16(lo, hh), s[kn]);
            }
        }
#pragma unroll
        for (int r = 0; r < 8; ++r) {
            const int qrow = qbase + r + hi * 8;
#pragma unroll
            for (int kn = 0; kn < 2; ++kn) {
                const int key = k0 + kn * 16 + lrow;
                if (key > qrow) s[kn][r] = -3.0e38f;
            }
            float mx = fmaxf(s[0][r], s[1][r]);
#pragma unroll
            for (int off = 8; off >= 1; off >>= 1)
                mx = fmaxf(mx, __shfl_xor(mx, off, 32));
            const float mnew  = fmaxf(m_i[r], mx);
            const float alpha = __expf(m_i[r] - mnew);
            const float p0 = __expf(s[0][r] - mnew);
            const float p1 = __expf(s[1][r] - mnew);
            s[0][r] = p0; s[1][r] = p1;
            float rsum = p0 + p1;
#pragma unroll
            for (int off = 8; off >= 1; off >>= 1)
                rsum += __shfl_xor(rsum, off, 32);
            l_i[r] = l_i[r] * alpha + rsum;
            m_i[r] = mnew;
#pragma unroll
            for (int dn = 0; dn < 4; ++dn) o[dn][r] *= alpha;
        }
#pragma unroll
        for (int r = 0; r < 8; ++r)
#pragma unroll
            for (int kn = 0; kn < 2; ++kn)
                sp[(r + hi * 8) * 32 + kn * 16 + lrow] = (_Float16)s[kn][r];
        const int ko = hi * 16;
        v8h plo = *(const v8h*)(sp + lrow * 32 + ko);
        v8h phi = *(const v8h*)(sp + lrow * 32 + ko + 8);
        v16h aP = cat16(plo, phi);
#pragma unroll
        for (int dn = 0; dn < 4; ++dn) {
            const int d   = dn * 16 + lrow;
            const int ko2 = hi * 16;
            v8h lo = *(const v8h*)(Vb + (size_t)d * T_DIM + k0 + ko2);
            v8h hh = *(const v8h*)(Vb + (size_t)d * T_DIM + k0 + ko2 + 8);
            o[dn] = wmma_f16(aP, cat16(lo, hh), o[dn]);
        }
    }

#pragma unroll
    for (int r = 0; r < 8; ++r) {
        const int t = qbase + r + hi * 8;
        const float inv = 1.0f / l_i[r];
#pragma unroll
        for (int dn = 0; dn < 4; ++dn) {
            const int d = dn * 16 + lrow;
            AO[((size_t)b * T_DIM + t) * C_DIM + h * D_HEAD + d] =
                (_Float16)(o[dn][r] * inv);
        }
    }
}

// ---------------------------------------------------------------------------
// Kernel 3: out = attn @ Wpt^T + b_proj (f32 out). A (f16) direct global,
// B (pre-transposed f16 weights) async-to-LDS, double-buffered.
// ---------------------------------------------------------------------------
__global__ void proj_gemm_kernel(const _Float16* __restrict__ AO,
                                 const _Float16* __restrict__ Wt,   // [1024][1024]
                                 const float* __restrict__ bias,
                                 float* __restrict__ out) {
    const int tid  = threadIdx.x;
    const int lane = tid & 31;
    const int wave = tid >> 5;
    const int wm   = wave >> 1;
    const int wn   = wave & 1;
    const int lrow = lane & 15;
    const int hi   = (lane >> 4) & 1;

    const int mbase = blockIdx.y * 64;
    const int nbase = blockIdx.x * 128;

    __shared__ __align__(16) _Float16 sB[2][128 * 32];
    const unsigned int sb_base = (unsigned int)(uintptr_t)&sB[0][0];

    v8f c[2][4];
#pragma unroll
    for (int mi = 0; mi < 2; ++mi)
#pragma unroll
        for (int ni = 0; ni < 4; ++ni) c[mi][ni] = v8f_zero();

    auto issue_tile = [&](int kt, int buf) {
#pragma unroll
        for (int j = 0; j < 4; ++j) {
            const int cidx = j * 128 + tid;
            const int n    = cidx >> 2;
            const int kh   = cidx & 3;
            const unsigned int lo = sb_base +
                (unsigned int)(buf * (128 * 32 * 2) + n * 64 + kh * 16);
            const _Float16* gp = Wt + ((size_t)(nbase + n) << 10) + kt * 32 + kh * 8;
            async_copy16(lo, gp);
        }
    };

    issue_tile(0, 0);

    for (int kt = 0; kt < C_DIM / 32; ++kt) {
        const int cur = kt & 1;
        __syncthreads();
        const bool more = (kt + 1) < (C_DIM / 32);
        if (more) {
            issue_tile(kt + 1, 1 - cur);
            wait_async_4();
        } else {
            wait_async_0();
        }
        __syncthreads();

        const int k0 = kt * 32;
        v16h a[2];
#pragma unroll
        for (int mi = 0; mi < 2; ++mi) {
            const int row = mbase + wm * 32 + mi * 16 + lrow;
            const int ko  = hi * 16;
            v8h lo = *(const v8h*)(AO + (size_t)row * C_DIM + k0 + ko);
            v8h hh = *(const v8h*)(AO + (size_t)row * C_DIM + k0 + ko + 8);
            a[mi] = cat16(lo, hh);
        }
#pragma unroll
        for (int ni = 0; ni < 4; ++ni) {
            const int nn = wn * 64 + ni * 16 + lrow;
            const int ko = hi * 16;
            v8h lo = *(const v8h*)(&sB[cur][nn * 32 + ko]);
            v8h hh = *(const v8h*)(&sB[cur][nn * 32 + ko + 8]);
            v16h bb = cat16(lo, hh);
#pragma unroll
            for (int mi = 0; mi < 2; ++mi)
                c[mi][ni] = wmma_f16(a[mi], bb, c[mi][ni]);
        }
    }

#pragma unroll
    for (int ni = 0; ni < 4; ++ni) {
        const int gn = nbase + wn * 64 + ni * 16 + lrow;
        const float bv = bias[gn];
#pragma unroll
        for (int mi = 0; mi < 2; ++mi)
#pragma unroll
            for (int r = 0; r < 8; ++r) {
                const int gm = mbase + wm * 32 + mi * 16 + r + hi * 8;
                out[(size_t)gm * C_DIM + gn] = c[mi][ni][r] + bv;
            }
    }
}

// ---------------------------------------------------------------------------
extern "C" void kernel_launch(void* const* d_in, const int* in_sizes, int n_in,
                              void* d_out, int out_size, void* d_ws, size_t ws_size,
                              hipStream_t stream) {
    (void)in_sizes; (void)n_in; (void)out_size; (void)ws_size;
    const float* x      = (const float*)d_in[0];
    const float* W_attn = (const float*)d_in[1];
    const float* b_attn = (const float*)d_in[2];
    const float* W_proj = (const float*)d_in[3];
    const float* b_proj = (const float*)d_in[4];
    float* out = (float*)d_out;

    // f16 workspace carve-up
    const size_t XH  = (size_t)B_DIM * T_DIM * C_DIM;              // 8388608
    const size_t HSZ = (size_t)B_DIM * N_HEADS * T_DIM * D_HEAD;   // 8388608
    const size_t WAT = (size_t)THREECC * C_DIM;                    // 3145728
    _Float16* ws  = (_Float16*)d_ws;
    _Float16* xh  = ws;
    _Float16* Qh  = xh + XH;
    _Float16* Kh  = Qh + HSZ;
    _Float16* Vh  = Kh + HSZ;
    _Float16* AO  = Vh + HSZ;
    _Float16* Wat = AO + HSZ;                  // [3072][1024] transposed
    _Float16* Wpt = Wat + WAT;                 // [1024][1024] transposed

    // 0) precision/layout pre-passes
    cvt_x_kernel<<<dim3((unsigned)(XH / 1024)), dim3(256), 0, stream>>>(x, xh);
    cvt_transpose_kernel<<<dim3(THREECC / 32, C_DIM / 32), dim3(256), 0, stream>>>(
        W_attn, Wat, THREECC);
    cvt_transpose_kernel<<<dim3(C_DIM / 32, C_DIM / 32), dim3(256), 0, stream>>>(
        W_proj, Wpt, C_DIM);

    // 1) QKV projection: M=8192, N=3072, K=1024
    qkv_gemm_kernel<<<dim3(THREECC / 128, (B_DIM * T_DIM) / 64), dim3(128), 0, stream>>>(
        xh, Wat, b_attn, Qh, Kh, Vh);

    // 2) causal flash attention
    flash_attn_kernel<<<dim3(T_DIM / 64, B_DIM * N_HEADS), dim3(128), 0, stream>>>(
        Qh, Kh, Vh, AO);

    // 3) output projection: M=8192, N=1024, K=1024
    proj_gemm_kernel<<<dim3(C_DIM / 128, (B_DIM * T_DIM) / 64), dim3(128), 0, stream>>>(
        AO, Wpt, b_proj, out);
}
